// TransformerDecoderLayer_33457795236439
// MI455X (gfx1250) — matrix operand order repaired and checked by reading, not compiled, not run on hardware
//
#include <hip/hip_runtime.h>
#include <math.h>

typedef __attribute__((ext_vector_type(16))) _Float16 v16h;
typedef __attribute__((ext_vector_type(8)))  _Float16 v8h;
typedef __attribute__((ext_vector_type(8)))  float    v8f;
typedef __attribute__((ext_vector_type(4)))  int      v4i;

#define D_MODEL 256
#define N_HEAD 8
#define HEAD_DIM 32
#define LQ 1000
#define BSZ 2
#define LV 87040

#define BM 128
#define BN 64

#if defined(__has_builtin)
#if __has_builtin(__builtin_amdgcn_global_load_async_to_lds_b128)
#define HAVE_ASYNC_LDS 1
#endif
#endif
#ifndef HAVE_ASYNC_LDS
#define HAVE_ASYNC_LDS 0
#endif

__device__ __forceinline__ v16h mkfrag(v8h a, v8h b) {
    v16h r;
#pragma unroll
    for (int i = 0; i < 8; ++i) { r[i] = a[i]; r[i + 8] = b[i]; }
    return r;
}

__device__ __forceinline__ v8h cvt8(float4 a, float4 b) {
    v8h r;
    r[0] = (_Float16)a.x; r[1] = (_Float16)a.y; r[2] = (_Float16)a.z; r[3] = (_Float16)a.w;
    r[4] = (_Float16)b.x; r[5] = (_Float16)b.y; r[6] = (_Float16)b.z; r[7] = (_Float16)b.w;
    return r;
}

__device__ __forceinline__ void async_cp16(const float* g, float* l) {
#if HAVE_ASYNC_LDS
    v4i* gp = (v4i*)g;      // strip const, retype to int4
    v4i* lp = (v4i*)l;
    __builtin_amdgcn_global_load_async_to_lds_b128(
        (__attribute__((address_space(1))) v4i*)gp,
        (__attribute__((address_space(3))) v4i*)lp,
        0, 0);
#else
    (void)g; (void)l;
#endif
}

__device__ __forceinline__ void async_wait0() {
#if HAVE_ASYNC_LDS
#if __has_builtin(__builtin_amdgcn_s_wait_asynccnt)
    __builtin_amdgcn_s_wait_asynccnt(0);
#else
    asm volatile("s_wait_asynccnt 0" ::: "memory");
#endif
#endif
}

// ---------------------------------------------------------------------------
// Generic tiled GEMM:  C[M,N] = act(A[M,K] @ W[N,K]^T + bias[N])
// fp32 inputs staged (async-to-LDS when available) -> fp16 LDS tiles ->
// V_WMMA_F32_16X16X32_F16, fp32 accumulate.
// Block = 256 threads (8 waves). Tile 128x64, K-step 32.
// ---------------------------------------------------------------------------
template <int ACT, bool HALF_OUT>
__global__ __launch_bounds__(256) void gemm_wmma(
    const float* __restrict__ A, int lda,
    const float* __restrict__ W, int ldw,
    const float* __restrict__ bias,
    void* __restrict__ Cout, int ldc,
    int M, int N, int K)
{
    __shared__ _Float16 Af[BM * 40];
    __shared__ _Float16 Wf[BN * 40];
#if HAVE_ASYNC_LDS
    __shared__ float Sa[BM * 32];
    __shared__ float Sw[BN * 32];
#endif

    const int tid  = threadIdx.x;
    const int wave = tid >> 5;            // 0..7
    const int lane = tid & 31;
    const int lrow = lane & 15;
    const int hsel = lane >> 4;
    const int m0 = blockIdx.y * BM;
    const int n0 = blockIdx.x * BN;

    v8f acc[4] = {};

    // A staging map: 128x32 elems, 256 thr -> 16 each
    const int ra = tid >> 1;              // row 0..127
    const int ca = (tid & 1) * 16;        // col base 0|16
    // W staging map: 64x32 elems, 256 thr -> 8 each
    const int rw = tid >> 2;              // row 0..63
    const int cw = (tid & 3) * 8;         // col base 0|8|16|24

    for (int k0 = 0; k0 < K; k0 += 32) {
        const bool fastA = (m0 + BM <= M) && (k0 + 32 <= K);
        const bool fastW = (n0 + BN <= N) && (k0 + 32 <= K);

#if HAVE_ASYNC_LDS
        if (fastA) {
            const float* src = A + (size_t)(m0 + ra) * lda + k0 + ca;
#pragma unroll
            for (int i = 0; i < 4; ++i)
                async_cp16(src + i * 4, &Sa[ra * 32 + ca + i * 4]);
        }
        if (fastW) {
            const float* src = W + (size_t)(n0 + rw) * ldw + k0 + cw;
#pragma unroll
            for (int i = 0; i < 2; ++i)
                async_cp16(src + i * 4, &Sw[rw * 32 + cw + i * 4]);
        }
#endif
        // slow (edge) paths write f16 tiles directly
        if (!fastA) {
            _Float16* dst = &Af[ra * 40 + ca];
            const int grow = m0 + ra;
            if (grow < M) {
                const float* src = A + (size_t)grow * lda + k0 + ca;
#pragma unroll
                for (int i = 0; i < 16; ++i) {
                    const int kk = k0 + ca + i;
                    dst[i] = (kk < K) ? (_Float16)src[i] : (_Float16)0.f;
                }
            } else {
#pragma unroll
                for (int i = 0; i < 16; ++i) dst[i] = (_Float16)0.f;
            }
        }
        if (!fastW) {
            _Float16* dst = &Wf[rw * 40 + cw];
            const int gn = n0 + rw;
            if (gn < N) {
                const float* src = W + (size_t)gn * ldw + k0 + cw;
#pragma unroll
                for (int i = 0; i < 8; ++i) {
                    const int kk = k0 + cw + i;
                    dst[i] = (kk < K) ? (_Float16)src[i] : (_Float16)0.f;
                }
            } else {
#pragma unroll
                for (int i = 0; i < 8; ++i) dst[i] = (_Float16)0.f;
            }
        }

#if HAVE_ASYNC_LDS
        if (fastA || fastW) async_wait0();
        __syncthreads();
        if (fastA) {
            const float4* s4 = (const float4*)&Sa[ra * 32 + ca];
            const float4 f0 = s4[0], f1 = s4[1], f2 = s4[2], f3 = s4[3];
            *(v8h*)&Af[ra * 40 + ca]     = cvt8(f0, f1);
            *(v8h*)&Af[ra * 40 + ca + 8] = cvt8(f2, f3);
        }
        if (fastW) {
            const float4* s4 = (const float4*)&Sw[rw * 32 + cw];
            *(v8h*)&Wf[rw * 40 + cw] = cvt8(s4[0], s4[1]);
        }
        __syncthreads();
#else
        if (fastA) {
            const float4* s4 = (const float4*)(A + (size_t)(m0 + ra) * lda + k0 + ca);
            __builtin_prefetch(s4 + 8, 0, 1);
            const float4 f0 = s4[0], f1 = s4[1], f2 = s4[2], f3 = s4[3];
            *(v8h*)&Af[ra * 40 + ca]     = cvt8(f0, f1);
            *(v8h*)&Af[ra * 40 + ca + 8] = cvt8(f2, f3);
        }
        if (fastW) {
            const float4* s4 = (const float4*)(W + (size_t)(n0 + rw) * ldw + k0 + cw);
            __builtin_prefetch(s4 + 8, 0, 1);
            *(v8h*)&Wf[rw * 40 + cw] = cvt8(s4[0], s4[1]);
        }
        __syncthreads();
#endif

        // ---- compute: wave owns output rows wave*16 .. wave*16+15 ----
        const int arow = wave * 16 + lrow;
        const int kbA  = hsel ? 8 : 0;
        const v16h af = mkfrag(*(const v8h*)&Af[arow * 40 + kbA],
                               *(const v8h*)&Af[arow * 40 + kbA + 16]);
        const int kbB = hsel ? 8 : 0;
#pragma unroll
        for (int j = 0; j < 4; ++j) {
            const int ncol = j * 16 + lrow;
            const v16h bf = mkfrag(*(const v8h*)&Wf[ncol * 40 + kbB],
                                   *(const v8h*)&Wf[ncol * 40 + kbB + 16]);
            acc[j] = __builtin_amdgcn_wmma_f32_16x16x32_f16(
                false, af, false, bf, (short)0, acc[j], false, false);
        }
        __syncthreads();
    }

    // ---- epilogue ----
#pragma unroll
    for (int j = 0; j < 4; ++j) {
        const int gcol = n0 + j * 16 + lrow;
        if (gcol >= N) continue;
        const float bv = bias[gcol];
#pragma unroll
        for (int v = 0; v < 8; ++v) {
            const int grow = m0 + wave * 16 + v + (hsel ? 8 : 0);
            if (grow >= M) continue;
            float val = acc[j][v] + bv;
            if (ACT == 1) val = fmaxf(val, 0.f);
            if (HALF_OUT)
                ((_Float16*)Cout)[(size_t)grow * ldc + gcol] = (_Float16)val;
            else
                ((float*)Cout)[(size_t)grow * ldc + gcol] = val;
        }
    }
}

// ---------------------------------------------------------------------------
// Flash-style MHA: one wave per (b, head, 16-query tile). HEAD_DIM=32.
// qkvh: (B, L, 768) fp16 = [q | k | v], each (H, 32) per token.
// ---------------------------------------------------------------------------
__global__ __launch_bounds__(32) void mha_flash(
    const _Float16* __restrict__ qkvh,
    float* __restrict__ outbuf,   // (B, L, 256)
    int L)
{
    __shared__ _Float16 Pl[16 * 40];
    __shared__ _Float16 Vt[32 * 40];   // Vt[dim][key]

    const int qt = blockIdx.x, h = blockIdx.y, b = blockIdx.z;
    const int lane = threadIdx.x;
    const int lrow = lane & 15;
    const int hsel = lane >> 4;

    const int qrow = qt * 16 + lrow;
    const int qc = (qrow < L) ? qrow : (L - 1);
    const _Float16* qbase = qkvh + ((size_t)b * L + qc) * 768 + h * 32;
    const int kbA = hsel ? 8 : 0;
    const v16h qa = mkfrag(*(const v8h*)(qbase + kbA),
                           *(const v8h*)(qbase + kbA + 16));

    float mrow[8], lsum[8];
#pragma unroll
    for (int v = 0; v < 8; ++v) { mrow[v] = -1e30f; lsum[v] = 0.f; }
    v8f o0 = {}, o1 = {};
    const float scale = 0.17677669529663687f;   // 1/sqrt(32)

    const int nChunks = (L + 31) / 32;
    for (int c = 0; c < nChunks; ++c) {
        const int kbase = c * 32;

        {   // stage V chunk transposed: lane = key-in-chunk
            const int key = kbase + lane;
            const int keyc = (key < L) ? key : (L - 1);
            const bool valid = key < L;
            const _Float16* vsrc = qkvh + ((size_t)b * L + keyc) * 768 + 512 + h * 32;
#pragma unroll
            for (int d = 0; d < 32; ++d)
                Vt[d * 40 + lane] = valid ? vsrc[d] : (_Float16)0.f;
        }

        v8f s[2];
#pragma unroll
        for (int sub = 0; sub < 2; ++sub) {
            const int kcol = kbase + sub * 16 + lrow;
            const int kc = (kcol < L) ? kcol : (L - 1);
            const _Float16* kp = qkvh + ((size_t)b * L + kc) * 768 + 256 + h * 32;
            const int kbB = hsel ? 8 : 0;
            const v16h bf = mkfrag(*(const v8h*)(kp + kbB),
                                   *(const v8h*)(kp + kbB + 16));
            v8f z = {};
            s[sub] = __builtin_amdgcn_wmma_f32_16x16x32_f16(
                false, qa, false, bf, (short)0, z, false, false);
#pragma unroll
            for (int v = 0; v < 8; ++v) {
                float sv = s[sub][v] * scale;
                if (kcol >= L) sv = -1e30f;
                s[sub][v] = sv;
            }
        }

        float mnew[8], rsc[8];
#pragma unroll
        for (int v = 0; v < 8; ++v) {
            float rm = fmaxf(s[0][v], s[1][v]);
            rm = fmaxf(rm, __shfl_xor(rm, 1));
            rm = fmaxf(rm, __shfl_xor(rm, 2));
            rm = fmaxf(rm, __shfl_xor(rm, 4));
            rm = fmaxf(rm, __shfl_xor(rm, 8));
            const float mn = fmaxf(mrow[v], rm);
            rsc[v] = __expf(mrow[v] - mn);
            mnew[v] = mn; mrow[v] = mn;
        }
#pragma unroll
        for (int sub = 0; sub < 2; ++sub) {
#pragma unroll
            for (int v = 0; v < 8; ++v) {
                const float p = __expf(s[sub][v] - mnew[v]);
                s[sub][v] = p;
                const int prow = v + (hsel ? 8 : 0);
                Pl[prow * 40 + sub * 16 + lrow] = (_Float16)p;
            }
        }
#pragma unroll
        for (int v = 0; v < 8; ++v) {
            float rs = s[0][v] + s[1][v];
            rs += __shfl_xor(rs, 1);
            rs += __shfl_xor(rs, 2);
            rs += __shfl_xor(rs, 4);
            rs += __shfl_xor(rs, 8);
            lsum[v] = lsum[v] * rsc[v] + rs;
            o0[v] *= rsc[v];
            o1[v] *= rsc[v];
        }
        __syncthreads();

        const v16h pa = mkfrag(*(const v8h*)&Pl[lrow * 40 + kbA],
                               *(const v8h*)&Pl[lrow * 40 + kbA + 16]);
        const int kbB = hsel ? 8 : 0;
        const v16h vb0 = mkfrag(*(const v8h*)&Vt[lrow * 40 + kbB],
                                *(const v8h*)&Vt[lrow * 40 + kbB + 16]);
        const v16h vb1 = mkfrag(*(const v8h*)&Vt[(16 + lrow) * 40 + kbB],
                                *(const v8h*)&Vt[(16 + lrow) * 40 + kbB + 16]);
        o0 = __builtin_amdgcn_wmma_f32_16x16x32_f16(false, pa, false, vb0, (short)0, o0, false, false);
        o1 = __builtin_amdgcn_wmma_f32_16x16x32_f16(false, pa, false, vb1, (short)0, o1, false, false);
        __syncthreads();
    }

#pragma unroll
    for (int v = 0; v < 8; ++v) {
        const int row = v + (hsel ? 8 : 0);
        const int q = qt * 16 + row;
        if (q < L) {
            const float inv = 1.0f / lsum[v];
            float* op = outbuf + ((size_t)b * L + q) * 256 + h * 32;
            op[lrow]      = o0[v] * inv;
            op[16 + lrow] = o1[v] * inv;
        }
    }
}

// ---------------------------------------------------------------------------
__global__ void add_kernel(const float* __restrict__ a, const float* __restrict__ b,
                           float* __restrict__ c, int n)
{
    int i = blockIdx.x * blockDim.x + threadIdx.x;
    if (i < n) c[i] = a[i] + b[i];
}

// ---------------------------------------------------------------------------
__global__ __launch_bounds__(256) void add_ln(
    const float* __restrict__ a, const float* __restrict__ r,
    const float* __restrict__ g, const float* __restrict__ be,
    float* __restrict__ out, int rows)
{
    const int wid = (blockIdx.x * blockDim.x + threadIdx.x) >> 5;
    const int lane = threadIdx.x & 31;
    if (wid >= rows) return;
    const float* pa = a + (size_t)wid * 256;
    const float* pr = r + (size_t)wid * 256;
    float vals[8], s = 0.f;
#pragma unroll
    for (int i = 0; i < 8; ++i) {
        const float v = pa[lane + 32 * i] + pr[lane + 32 * i];
        vals[i] = v; s += v;
    }
#pragma unroll
    for (int d = 16; d; d >>= 1) s += __shfl_xor(s, d);
    const float mean = s * (1.f / 256.f);
    float vs = 0.f;
#pragma unroll
    for (int i = 0; i < 8; ++i) { const float d = vals[i] - mean; vs += d * d; }
#pragma unroll
    for (int d = 16; d; d >>= 1) vs += __shfl_xor(vs, d);
    const float rstd = rsqrtf(vs * (1.f / 256.f) + 1e-5f);
    float* po = out + (size_t)wid * 256;
#pragma unroll
    for (int i = 0; i < 8; ++i) {
        const int c = lane + 32 * i;
        po[c] = (vals[i] - mean) * rstd * g[c] + be[c];
    }
}

// ---------------------------------------------------------------------------
__global__ __launch_bounds__(256) void softmax100(float* __restrict__ attnw, int rows)
{
    const int wid = (blockIdx.x * blockDim.x + threadIdx.x) >> 5;
    const int lane = threadIdx.x & 31;
    if (wid >= rows) return;
    const int bl = wid >> 3, h = wid & 7;
    float* p = attnw + (size_t)bl * 800 + h * 100;
    float v0 = p[lane], v1 = p[32 + lane], v2 = p[64 + lane];
    const bool t = (96 + lane) < 100;
    float v3 = t ? p[96 + lane] : -1e30f;
    float mx = fmaxf(fmaxf(v0, v1), fmaxf(v2, v3));
#pragma unroll
    for (int d = 16; d; d >>= 1) mx = fmaxf(mx, __shfl_xor(mx, d));
    float e0 = __expf(v0 - mx), e1 = __expf(v1 - mx), e2 = __expf(v2 - mx);
    float e3 = t ? __expf(v3 - mx) : 0.f;
    float s = e0 + e1 + e2 + e3;
#pragma unroll
    for (int d = 16; d; d >>= 1) s += __shfl_xor(s, d);
    const float inv = 1.f / s;
    p[lane] = e0 * inv; p[32 + lane] = e1 * inv; p[64 + lane] = e2 * inv;
    if (t) p[96 + lane] = e3 * inv;
}

// ---------------------------------------------------------------------------
__global__ __launch_bounds__(256) void box_sample(
    const _Float16* __restrict__ valueh,  // (B, LV, 256)
    const float* __restrict__ attnw,      // (B, L, 800) softmaxed
    const float* __restrict__ boxoff,     // (B, L, 160)
    const float* __restrict__ refw,       // (B, L, 10)
    float* __restrict__ samp,             // (B, L, 256)
    int L)
{
    const int wid = (blockIdx.x * blockDim.x + threadIdx.x) >> 5;
    const int lane = threadIdx.x & 31;
    if (wid >= BSZ * L * N_HEAD) return;
    const int h = wid & 7;
    const int bl = wid >> 3;
    const int l = bl % L;
    const int b = bl / L;

    const float* rw = refw + ((size_t)b * L + l) * 10;
    const float r0 = rw[0], r1 = rw[1], rww = rw[3], rhh = rw[4], rang = rw[6];
    const float* off = boxoff + ((size_t)b * L + l) * 160 + h * 20;
    const float* aw  = attnw + ((size_t)b * L + l) * 800 + h * 100;
    const _Float16* vb = valueh + (size_t)b * LV * 256;

    const int HW[4] = {256, 128, 64, 32};
    const int ST[4] = {0, 65536, 81920, 86016};
    float acc = 0.f;

#pragma unroll
    for (int lvl = 0; lvl < 4; ++lvl) {
        const float o0 = off[lvl * 5 + 0], o1 = off[lvl * 5 + 1];
        const float o2 = off[lvl * 5 + 2], o3 = off[lvl * 5 + 3];
        const float o4 = off[lvl * 5 + 4];
        const float ang = (rang + o4 * (1.f / 16.f)) * 6.283185307179586f;
        float sn, cs;
        __sincosf(ang, &sn, &cs);
        const float cx = r0 + o0 * 0.125f * rww;
        const float cy = r1 + o1 * 0.125f * rhh;
        const float sw = fmaxf(rww + o2 * 0.125f * rww, 0.f);
        const float sh = fmaxf(rhh + o3 * 0.125f * rhh, 0.f);
        const int Wl = HW[lvl], Hl = HW[lvl], st = ST[lvl];

        for (int p = 0; p < 25; ++p) {
            const float kx = (float)((p % 5) - 2) * 0.2f;
            const float ky = (float)((p / 5) - 2) * 0.2f;
            const float g0 = kx * sw, g1 = ky * sh;
            const float gx = cs * g0 - sn * g1;
            const float gy = sn * g0 + cs * g1;
            const float x = (cx + gx) * (float)Wl - 0.5f;
            const float y = (cy + gy) * (float)Hl - 0.5f;
            const float x0f = floorf(x), y0f = floorf(y);
            const float fx = x - x0f, fy = y - y0f;
            const int x0 = (int)x0f, y0 = (int)y0f;
            const float a = aw[lvl * 25 + p];
            float sval = 0.f;
#pragma unroll
            for (int dy = 0; dy < 2; ++dy)
#pragma unroll
                for (int dx = 0; dx < 2; ++dx) {
                    const int xi = x0 + dx, yi = y0 + dy;
                    if (xi >= 0 && xi < Wl && yi >= 0 && yi < Hl) {
                        const float w = (dx ? fx : 1.f - fx) * (dy ? fy : 1.f - fy);
                        const float v = (float)vb[((size_t)(st + yi * Wl + xi)) * 256 + h * 32 + lane];
                        sval += w * v;
                    }
                }
            acc += a * sval;
        }
    }
    samp[((size_t)b * L + l) * 256 + h * 32 + lane] = acc;
}

// ---------------------------------------------------------------------------
extern "C" void kernel_launch(void* const* d_in, const int* in_sizes, int n_in,
                              void* d_out, int out_size, void* d_ws, size_t ws_size,
                              hipStream_t stream)
{
    const float* query  = (const float*)d_in[0];
    const float* memory = (const float*)d_in[1];
    const float* refw   = (const float*)d_in[2];
    const float* w_in     = (const float*)d_in[5];
    const float* b_in     = (const float*)d_in[6];
    const float* w_sa_out = (const float*)d_in[7];
    const float* b_sa_out = (const float*)d_in[8];
    const float* w_pos1   = (const float*)d_in[9];
    const float* b_pos1   = (const float*)d_in[10];
    const float* w_pos2   = (const float*)d_in[11];
    const float* b_pos2   = (const float*)d_in[12];
    const float* w_pos3   = (const float*)d_in[13];
    const float* b_pos3   = (const float*)d_in[14];
    const float* w_value  = (const float*)d_in[15];
    const float* b_value  = (const float*)d_in[16];
    const float* w_attn   = (const float*)d_in[17];
    const float* b_attn   = (const float*)d_in[18];
    const float* w_box    = (const float*)d_in[19];
    const float* b_box    = (const float*)d_in[20];
    const float* w_bo     = (const float*)d_in[21];
    const float* b_bo     = (const float*)d_in[22];
    const float* w_ff1    = (const float*)d_in[23];
    const float* b_ff1    = (const float*)d_in[24];
    const float* w_ff2    = (const float*)d_in[25];
    const float* b_ff2    = (const float*)d_in[26];
    const float* g1  = (const float*)d_in[27];
    const float* be1 = (const float*)d_in[28];
    const float* g2  = (const float*)d_in[29];
    const float* be2 = (const float*)d_in[30];
    const float* g3  = (const float*)d_in[31];
    const float* be3 = (const float*)d_in[32];

    const int ROWS = BSZ * LQ;            // 2000
    const int NE   = ROWS * D_MODEL;      // 512000

    char* p = (char*)d_ws;
    auto alloc = [&](size_t bytes) -> char* {
        char* r = p; p += (bytes + 255) & ~(size_t)255; return r;
    };
    float* htmpA   = (float*)alloc((size_t)NE * 4);
    float* htmpB   = (float*)alloc((size_t)NE * 4);
    float* qpos    = (float*)alloc((size_t)NE * 4);
    float* qbuf    = (float*)alloc((size_t)NE * 4);
    _Float16* qkvh = (_Float16*)alloc((size_t)ROWS * 768 * 2);
    float* sa_attn = (float*)alloc((size_t)NE * 4);
    float* sa_out  = (float*)alloc((size_t)NE * 4);
    float* x1      = (float*)alloc((size_t)NE * 4);
    float* boxin   = (float*)alloc((size_t)NE * 4);
    float* attnw   = (float*)alloc((size_t)ROWS * 800 * 4);
    float* boxoff  = (float*)alloc((size_t)ROWS * 160 * 4);
    _Float16* valueh = (_Float16*)alloc((size_t)BSZ * LV * 256 * 2);
    float* samp    = (float*)alloc((size_t)NE * 4);
    float* boxout  = (float*)alloc((size_t)NE * 4);
    float* x2      = (float*)alloc((size_t)NE * 4);
    float* ffh     = (float*)alloc((size_t)ROWS * 1024 * 4);
    float* ff2     = (float*)alloc((size_t)NE * 4);

    auto ggrid = [](int M, int N) { return dim3((N + BN - 1) / BN, (M + BM - 1) / BM); };
    const dim3 blk(256);

    // 1) query_pos = mlp3(ref_windows)
    gemm_wmma<1, false><<<ggrid(ROWS, 256), blk, 0, stream>>>(refw, 10, w_pos1, 10, b_pos1, htmpA, 256, ROWS, 256, 10);
    gemm_wmma<1, false><<<ggrid(ROWS, 256), blk, 0, stream>>>(htmpA, 256, w_pos2, 256, b_pos2, htmpB, 256, ROWS, 256, 256);
    gemm_wmma<0, false><<<ggrid(ROWS, 256), blk, 0, stream>>>(htmpB, 256, w_pos3, 256, b_pos3, qpos, 256, ROWS, 256, 256);

    // 2) q = query + query_pos
    add_kernel<<<(NE + 255) / 256, 256, 0, stream>>>(query, qpos, qbuf, NE);

    // 3) MHA
    gemm_wmma<0, true><<<ggrid(ROWS, 512), blk, 0, stream>>>(qbuf, 256, w_in, 256, b_in, qkvh, 768, ROWS, 512, 256);
    gemm_wmma<0, true><<<ggrid(ROWS, 256), blk, 0, stream>>>(query, 256, w_in + (size_t)512 * 256, 256, b_in + 512, qkvh + 512, 768, ROWS, 256, 256);
    mha_flash<<<dim3((LQ + 15) / 16, N_HEAD, BSZ), 32, 0, stream>>>(qkvh, sa_attn, LQ);
    gemm_wmma<0, false><<<ggrid(ROWS, 256), blk, 0, stream>>>(sa_attn, 256, w_sa_out, 256, b_sa_out, sa_out, 256, ROWS, 256, 256);

    // 4) x1 = LN(query + sa_out)
    add_ln<<<(ROWS * 32 + 255) / 256, 256, 0, stream>>>(sa_out, query, g1, be1, x1, ROWS);

    // 5) box attention
    add_kernel<<<(NE + 255) / 256, 256, 0, stream>>>(x1, qpos, boxin, NE);
    gemm_wmma<0, false><<<ggrid(ROWS, 800), blk, 0, stream>>>(boxin, 256, w_attn, 256, b_attn, attnw, 800, ROWS, 800, 256);
    softmax100<<<(ROWS * 8 * 32 + 255) / 256, 256, 0, stream>>>(attnw, ROWS * 8);
    gemm_wmma<0, false><<<ggrid(ROWS, 160), blk, 0, stream>>>(boxin, 256, w_box, 256, b_box, boxoff, 160, ROWS, 160, 256);
    gemm_wmma<0, true><<<ggrid(BSZ * LV, 256), blk, 0, stream>>>(memory, 256, w_value, 256, b_value, valueh, 256, BSZ * LV, 256, 256);
    box_sample<<<(ROWS * 8 * 32 + 255) / 256, 256, 0, stream>>>(valueh, attnw, boxoff, refw, samp, LQ);
    gemm_wmma<0, false><<<ggrid(ROWS, 256), blk, 0, stream>>>(samp, 256, w_bo, 256, b_bo, boxout, 256, ROWS, 256, 256);

    // 6) x2 = LN(x1 + box)
    add_ln<<<(ROWS * 32 + 255) / 256, 256, 0, stream>>>(boxout, x1, g2, be2, x2, ROWS);

    // 7) FFN + final LN -> d_out
    gemm_wmma<1, false><<<ggrid(ROWS, 1024), blk, 0, stream>>>(x2, 256, w_ff1, 256, b_ff1, ffh, 1024, ROWS, 1024, 256);
    gemm_wmma<0, false><<<ggrid(ROWS, 256), blk, 0, stream>>>(ffh, 1024, w_ff2, 1024, b_ff2, ff2, 256, ROWS, 256, 1024);
    add_ln<<<(ROWS * 32 + 255) / 256, 256, 0, stream>>>(ff2, x2, g3, be3, (float*)d_out, ROWS);
}
